// MultiHeadAttention_48679159332914
// MI455X (gfx1250) — matrix operand order repaired and checked by reading, not compiled, not run on hardware
//
#include <hip/hip_runtime.h>
#include <hip/hip_bf16.h>

// Problem constants (match reference)
#define B_  8
#define D_  256
#define N_  2048
#define H_  4
#define HD_ 64

typedef _Float16 v16h __attribute__((ext_vector_type(16)));
typedef float    v8f  __attribute__((ext_vector_type(8)));
typedef unsigned int u32x4 __attribute__((ext_vector_type(4)));
typedef int          i32x8 __attribute__((ext_vector_type(8)));
typedef int          i32x4 __attribute__((ext_vector_type(4)));
typedef __attribute__((address_space(3))) char lds_char;

#define TDM_AVAILABLE (__has_builtin(__builtin_amdgcn_tensor_load_to_lds) && \
                       __has_builtin(__builtin_amdgcn_s_wait_tensorcnt))

// ---- WMMA fragment index helpers (CDNA5 ISA 7.12.2, wave32) ----
// A-matrix 16x32 f16: lane L holds row M = L%16; VGPR v holds
// K = (v/4)*16 + (L/16)*8 + (v%4)*2  (+1 in high half)
__device__ __forceinline__ int a_frag_k(int lane, int v) {
  return ((v >> 2) << 4) + ((lane >> 4) << 3) + ((v & 3) << 1);
}
// B-matrix 32x16 f16: lane L holds col N = L%16; VGPR v holds
// K = (v/4)*16 + (L/16)*8 + (v%4)*2  (+1 in high half)
__device__ __forceinline__ int b_frag_k(int lane, int v) {
  return ((v >> 2) << 4) + ((lane >> 4) << 3) + ((v & 3) << 1);
}

// Load A fragment from [M][K] row-major f16 storage (K pairs contiguous).
__device__ __forceinline__ v16h load_a_frag(const _Float16* base, int rowStride, int lane) {
  v16h f;
  const _Float16* p = base + (size_t)(lane & 15) * rowStride;
#pragma unroll
  for (int v = 0; v < 8; ++v) {
    int k = a_frag_k(lane, v);
    f[2 * v]     = p[k];
    f[2 * v + 1] = p[k + 1];
  }
  return f;
}
// Load B fragment from [N][K] row-major f16 storage (i.e. B^T; column-major B).
__device__ __forceinline__ v16h load_b_frag(const _Float16* base, int rowStride, int lane) {
  v16h f;
  const _Float16* p = base + (size_t)(lane & 15) * rowStride;
#pragma unroll
  for (int v = 0; v < 8; ++v) {
    int k = b_frag_k(lane, v);
    f[2 * v]     = p[k];
    f[2 * v + 1] = p[k + 1];
  }
  return f;
}
// Load B fragment from [K][N] row-major f16 storage (natural B layout).
__device__ __forceinline__ v16h load_b_frag_rm(const _Float16* base, int rowStride, int lane) {
  v16h f;
  const _Float16* p = base + (lane & 15);
#pragma unroll
  for (int v = 0; v < 8; ++v) {
    int k = b_frag_k(lane, v);
    f[2 * v]     = p[(size_t)k * rowStride];
    f[2 * v + 1] = p[(size_t)(k + 1) * rowStride];
  }
  return f;
}

__device__ __forceinline__ v8f wmma16(v16h a, v16h b, v8f c) {
  // D = A*B + C, f32 accumulate
  return __builtin_amdgcn_wmma_f32_16x16x32_f16(false, a, false, b, (short)0, c, false, false);
}

// Reductions within each 16-lane half of a wave32 (xor masks 1..8 never cross bit 4)
__device__ __forceinline__ float red_max16(float v) {
#pragma unroll
  for (int o = 8; o >= 1; o >>= 1) v = fmaxf(v, __shfl_xor(v, o, 32));
  return v;
}
__device__ __forceinline__ float red_sum16(float v) {
#pragma unroll
  for (int o = 8; o >= 1; o >>= 1) v += __shfl_xor(v, o, 32);
  return v;
}

// =====================================================================
// Kernel 1: input projection GEMM.
//   dst_f16[b][h][n][hd] = ((sum_d W[d][o] * X[b][d][n]) + bias[o]) * scale
//   with o = hd*H + h  (reference reshape splits D as (head_dim, heads))
// Block: 256 thr = 8 waves (4 M x 2 N), tile 128(n) x 64(o), K step 32.
// =====================================================================
__global__ __launch_bounds__(256) void mha_proj_kernel(
    const float* __restrict__ X, const float* __restrict__ W,
    const float* __restrict__ bias, _Float16* __restrict__ dst, float scale) {
  const int b  = blockIdx.z;
  const int n0 = blockIdx.x * 128;
  const int o0 = blockIdx.y * 64;
  const int t    = threadIdx.x;
  const int lane = t & 31;
  const int w    = t >> 5;
  const int wm = w & 3;   // M (n) quadrant
  const int wn = w >> 2;  // N (o) half
  const int lane15 = lane & 15;
  const int half   = lane >> 4;

  __shared__ __align__(16) _Float16 As[128 * 40];  // [n_local][k], stride 40
  __shared__ __align__(16) _Float16 Bs[64 * 40];   // [o_local][k], stride 40

  const float* Xb = X + (size_t)b * D_ * N_;

  v8f c[2][2] = {{{}, {}}, {{}, {}}};

  for (int k0 = 0; k0 < D_; k0 += 32) {
    __syncthreads();
    // Stage A tile: X[d=k0..k0+32][n=n0..n0+128] -> As[n][d] (f32->f16 transpose)
#pragma unroll
    for (int i = 0; i < 4; ++i) {
      int f    = t + i * 256;   // 0..1023
      int row  = f >> 5;        // d_local 0..31
      int col4 = f & 31;        // n group of 4
      const float4 xv = *(const float4*)(Xb + (size_t)(k0 + row) * N_ + n0 + col4 * 4);
      int nl = col4 * 4;
      As[(nl + 0) * 40 + row] = (_Float16)xv.x;
      As[(nl + 1) * 40 + row] = (_Float16)xv.y;
      As[(nl + 2) * 40 + row] = (_Float16)xv.z;
      As[(nl + 3) * 40 + row] = (_Float16)xv.w;
    }
    // Stage B tile: W[d=k0..+32][o=o0..+64] -> Bs[o][d]
#pragma unroll
    for (int i = 0; i < 2; ++i) {
      int f    = t + i * 256;   // 0..511
      int row  = f >> 4;        // d_local 0..31
      int col4 = f & 15;        // o group of 4
      const float4 wv = *(const float4*)(W + (size_t)(k0 + row) * D_ + o0 + col4 * 4);
      int ol = col4 * 4;
      Bs[(ol + 0) * 40 + row] = (_Float16)wv.x;
      Bs[(ol + 1) * 40 + row] = (_Float16)wv.y;
      Bs[(ol + 2) * 40 + row] = (_Float16)wv.z;
      Bs[(ol + 3) * 40 + row] = (_Float16)wv.w;
    }
    if (k0 + 32 < D_) __builtin_prefetch(Xb + (size_t)(k0 + 32) * N_ + n0, 0, 1);
    __syncthreads();

    v16h a[2], bb[2];
    a[0]  = load_a_frag(As + (wm * 32 + 0)  * 40, 40, lane);
    a[1]  = load_a_frag(As + (wm * 32 + 16) * 40, 40, lane);
    bb[0] = load_b_frag(Bs + (wn * 32 + 0)  * 40, 40, lane);
    bb[1] = load_b_frag(Bs + (wn * 32 + 16) * 40, 40, lane);
#pragma unroll
    for (int mi = 0; mi < 2; ++mi)
#pragma unroll
      for (int nj = 0; nj < 2; ++nj) c[mi][nj] = wmma16(a[mi], bb[nj], c[mi][nj]);
  }

  // Epilogue: + bias, * scale, f16 -> dst[b][h][n][hd]
#pragma unroll
  for (int mi = 0; mi < 2; ++mi) {
#pragma unroll
    for (int nj = 0; nj < 2; ++nj) {
      int o  = o0 + wn * 32 + nj * 16 + lane15;
      int hh = o & (H_ - 1);
      int hd = o >> 2;
      float bval = bias[o];
      _Float16* drow = dst + ((size_t)(b * H_ + hh) * N_) * HD_ + hd;
#pragma unroll
      for (int r = 0; r < 8; ++r) {
        int n = n0 + wm * 32 + mi * 16 + r + 8 * half;
        drow[(size_t)n * HD_] = (_Float16)((c[mi][nj][r] + bval) * scale);
      }
    }
  }
}

// =====================================================================
// Kernel 2: flash attention per (b, h, 128-query tile).
// 8 waves; each wave owns 16 query rows. 64-key K/V chunks are DMA'd
// into LDS by the Tensor Data Mover (row-major, pad to stride 66 f16).
// =====================================================================
__global__ __launch_bounds__(256) void mha_attn_kernel(
    const _Float16* __restrict__ Qws, const _Float16* __restrict__ Kws,
    const _Float16* __restrict__ Vws, _Float16* __restrict__ Xo) {
  const int b  = blockIdx.z;
  const int h  = blockIdx.y;
  const int n0 = blockIdx.x * 128;
  const int t    = threadIdx.x;
  const int lane = t & 31;
  const int w    = t >> 5;
  const int lane15 = lane & 15;
  const int half   = lane >> 4;

  __shared__ __align__(16) _Float16 Ks[64 * 66];       // [m][d], stride 66 (TDM pad)
  __shared__ __align__(16) _Float16 Vr[64 * 66];       // [m][e], stride 66 (TDM pad)
  __shared__ __align__(16) _Float16 Ps[8 * 16 * 72];   // per-wave P strip [16][72]

  const size_t bh = (size_t)(b * H_ + h) * N_;
  const _Float16* Qp = Qws + bh * HD_;
  const _Float16* Kp = Kws + bh * HD_;
  const _Float16* Vp = Vws + bh * HD_;

  // This wave's Q fragments (16 rows x 64 head dims, already scaled by 1/8)
  const int qrow = n0 + w * 16;
  v16h aq[2];
  aq[0] = load_a_frag(Qp + (size_t)qrow * HD_ + 0,  HD_, lane);
  aq[1] = load_a_frag(Qp + (size_t)qrow * HD_ + 32, HD_, lane);

  v8f oacc[4] = {{}, {}, {}, {}};
  float mrun[8], lrun[8];
#pragma unroll
  for (int r = 0; r < 8; ++r) { mrun[r] = -1e30f; lrun[r] = 0.0f; }

  _Float16* Pw = Ps + w * 16 * 72;

#if TDM_AVAILABLE
  // TDM D# (CDNA5 ISA): 2D f16 tensor, 64x64 tile,
  // dim0 (contiguous) = 64 elems, dim0_stride = 64, pad 1 DWORD per 128B row
  // (pad_enable=1, pad_interval=4 -> 16*8B, pad_amount=0 -> 1 DWORD) so the
  // LDS destination has stride 66 halves (33 DWORDs: conflict-free vs 64 banks).
  const unsigned ks_off = (unsigned)(size_t)(lds_char*)Ks;
  const unsigned vr_off = (unsigned)(size_t)(lds_char*)Vr;
  const i32x8 g1t = { 0x01110000,        // wg_mask=0, data_size=1(2B), pad cfg
                      0x00400000,        // tensor_dim0 = 64 (bits 79:48 low)
                      0x00400000,        // tensor_dim1 = 64 (bits 111:80 low)
                      0x00400000,        // tile_dim0 = 64  (bits 127:112)
                      64,                // tile_dim1 = 64, tile_dim2 = 0
                      64,                // tensor_dim0_stride = 64
                      0, 0 };            // dim0_stride hi, dim1_stride = 0
  const i32x4 gz4 = {0, 0, 0, 0};
  const i32x8 gz8 = {0, 0, 0, 0, 0, 0, 0, 0};
#endif

  for (int c0 = 0; c0 < N_; c0 += 64) {
    __syncthreads();  // previous chunk fully consumed before overwrite
#if TDM_AVAILABLE
    if (w == 0) {
      unsigned long long ka = (unsigned long long)(Kp + (size_t)c0 * HD_);
      unsigned long long va = (unsigned long long)(Vp + (size_t)c0 * HD_);
      u32x4 g0k = { 1u, ks_off, (unsigned)ka,
                    (unsigned)((ka >> 32) & 0x01FFFFFFull) | 0x80000000u };  // type=2
      u32x4 g0v = { 1u, vr_off, (unsigned)va,
                    (unsigned)((va >> 32) & 0x01FFFFFFull) | 0x80000000u };
      // 6-arg toolchain variant: (g0, g1, g2, g3, g4, cpol)
      __builtin_amdgcn_tensor_load_to_lds(g0k, g1t, gz4, gz4, gz8, 0);
      __builtin_amdgcn_tensor_load_to_lds(g0v, g1t, gz4, gz4, gz8, 0);
      __builtin_amdgcn_s_wait_tensorcnt(0);
    }
#else
    {  // fallback cooperative copy (also what the host pass parses)
      const int m = t >> 2, part = t & 3;
      const _Float16* ksrc = Kp + (size_t)(c0 + m) * HD_ + part * 16;
      const _Float16* vsrc = Vp + (size_t)(c0 + m) * HD_ + part * 16;
#pragma unroll
      for (int j = 0; j < 16; ++j) {
        Ks[m * 66 + part * 16 + j] = ksrc[j];
        Vr[m * 66 + part * 16 + j] = vsrc[j];
      }
    }
#endif
    __syncthreads();

    // S = Q * K^T : 16 x 64 strip (4 N-fragments, K=64 in 2 steps)
    v8f s[4];
#pragma unroll
    for (int nj = 0; nj < 4; ++nj) {
      v8f acc = {};
#pragma unroll
      for (int kt = 0; kt < 2; ++kt) {
        v16h bk = load_b_frag(Ks + (nj * 16) * 66 + kt * 32, 66, lane);
        acc = wmma16(aq[kt], bk, acc);
      }
      s[nj] = acc;
    }

    // Online softmax per row (rows r+8*half live in 16-lane halves)
#pragma unroll
    for (int r = 0; r < 8; ++r) {
      float mx = fmaxf(fmaxf(s[0][r], s[1][r]), fmaxf(s[2][r], s[3][r]));
      mx = red_max16(mx);
      float mnew  = fmaxf(mrun[r], mx);
      float alpha = __expf(mrun[r] - mnew);
      float ssum  = 0.0f;
      int row = r + 8 * half;
#pragma unroll
      for (int nj = 0; nj < 4; ++nj) {
        float p = __expf(s[nj][r] - mnew);
        ssum += p;
        Pw[row * 72 + nj * 16 + lane15] = (_Float16)p;
      }
      ssum = red_sum16(ssum);
      lrun[r] = lrun[r] * alpha + ssum;
      mrun[r] = mnew;
#pragma unroll
      for (int ej = 0; ej < 4; ++ej) oacc[ej][r] *= alpha;
    }

    // O += P * V (P re-read in A layout from this wave's LDS strip,
    // V fragments built from the row-major TDM-staged tile)
    v16h ap[2];
    ap[0] = load_a_frag(Pw + 0,  72, lane);
    ap[1] = load_a_frag(Pw + 32, 72, lane);
#pragma unroll
    for (int ej = 0; ej < 4; ++ej) {
#pragma unroll
      for (int kt = 0; kt < 2; ++kt) {
        v16h bv = load_b_frag_rm(Vr + (size_t)(kt * 32) * 66 + ej * 16, 66, lane);
        oacc[ej] = wmma16(ap[kt], bv, oacc[ej]);
      }
    }
  }

  // Normalize and scatter: x[b][n][d = e*H + h] as f16
#pragma unroll
  for (int ej = 0; ej < 4; ++ej) {
    int e = ej * 16 + lane15;
#pragma unroll
    for (int r = 0; r < 8; ++r) {
      int n = qrow + r + 8 * half;
      float val = oacc[ej][r] / lrun[r];
      Xo[((size_t)b * N_ + n) * D_ + e * H_ + h] = (_Float16)val;
    }
  }
}

// =====================================================================
// Kernel 3: output projection.
//   out[b][o][n] = sum_d Wm[d][o] * x[b][n][d] + bm[o]   (fp32 out)
// A-fragments load x (f16, [n][d]) straight from global.
// =====================================================================
__global__ __launch_bounds__(256) void mha_out_kernel(
    const _Float16* __restrict__ Xi, const float* __restrict__ Wm,
    const float* __restrict__ bm, float* __restrict__ out) {
  const int b  = blockIdx.z;
  const int n0 = blockIdx.x * 128;
  const int o0 = blockIdx.y * 64;
  const int t    = threadIdx.x;
  const int lane = t & 31;
  const int w    = t >> 5;
  const int wm = w & 3;
  const int wn = w >> 2;
  const int lane15 = lane & 15;
  const int half   = lane >> 4;

  __shared__ __align__(16) _Float16 Bs[64 * 40];  // [o_local][k]

  const _Float16* Xb = Xi + (size_t)b * N_ * D_;

  v8f c[2][2] = {{{}, {}}, {{}, {}}};

  for (int k0 = 0; k0 < D_; k0 += 32) {
    __syncthreads();
#pragma unroll
    for (int i = 0; i < 2; ++i) {
      int f    = t + i * 256;
      int row  = f >> 4;
      int col4 = f & 15;
      const float4 wv = *(const float4*)(Wm + (size_t)(k0 + row) * D_ + o0 + col4 * 4);
      int ol = col4 * 4;
      Bs[(ol + 0) * 40 + row] = (_Float16)wv.x;
      Bs[(ol + 1) * 40 + row] = (_Float16)wv.y;
      Bs[(ol + 2) * 40 + row] = (_Float16)wv.z;
      Bs[(ol + 3) * 40 + row] = (_Float16)wv.w;
    }
    __syncthreads();

    v16h a[2], bb[2];
    a[0]  = load_a_frag(Xb + (size_t)(n0 + wm * 32 + 0)  * D_ + k0, D_, lane);
    a[1]  = load_a_frag(Xb + (size_t)(n0 + wm * 32 + 16) * D_ + k0, D_, lane);
    bb[0] = load_b_frag(Bs + (wn * 32 + 0)  * 40, 40, lane);
    bb[1] = load_b_frag(Bs + (wn * 32 + 16) * 40, 40, lane);
#pragma unroll
    for (int mi = 0; mi < 2; ++mi)
#pragma unroll
      for (int nj = 0; nj < 2; ++nj) c[mi][nj] = wmma16(a[mi], bb[nj], c[mi][nj]);
  }

#pragma unroll
  for (int mi = 0; mi < 2; ++mi) {
#pragma unroll
    for (int nj = 0; nj < 2; ++nj) {
      int o = o0 + wn * 32 + nj * 16 + lane15;
      float bval = bm[o];
      float* obase = out + ((size_t)b * D_ + o) * N_;
#pragma unroll
      for (int r = 0; r < 8; ++r) {
        int n = n0 + wm * 32 + mi * 16 + r + 8 * half;
        obase[n] = c[mi][nj][r] + bval;
      }
    }
  }
}

extern "C" void kernel_launch(void* const* d_in, const int* in_sizes, int n_in,
                              void* d_out, int out_size, void* d_ws, size_t ws_size,
                              hipStream_t stream) {
  (void)in_sizes; (void)n_in; (void)out_size; (void)ws_size;
  const float* query = (const float*)d_in[0];
  const float* key   = (const float*)d_in[1];
  const float* value = (const float*)d_in[2];
  const float* Wq = (const float*)d_in[3];
  const float* bq = (const float*)d_in[4];
  const float* Wk = (const float*)d_in[5];
  const float* bk = (const float*)d_in[6];
  const float* Wv = (const float*)d_in[7];
  const float* bv = (const float*)d_in[8];
  const float* Wm = (const float*)d_in[9];
  const float* bm = (const float*)d_in[10];

  // Workspace: q/k/v f16 [B][H][N][64] + x f16 [B][N][D] = 32 MB total
  const size_t per = (size_t)B_ * H_ * N_ * HD_;
  _Float16* qws = (_Float16*)d_ws;
  _Float16* kws = qws + per;
  _Float16* vws = kws + per;
  _Float16* xws = vws + per;

  dim3 blk(256);
  dim3 gproj(N_ / 128, D_ / 64, B_);
  mha_proj_kernel<<<gproj, blk, 0, stream>>>(query, Wq, bq, qws, 0.125f);  // 1/sqrt(64)
  mha_proj_kernel<<<gproj, blk, 0, stream>>>(key,   Wk, bk, kws, 1.0f);
  mha_proj_kernel<<<gproj, blk, 0, stream>>>(value, Wv, bv, vws, 1.0f);

  dim3 gattn(N_ / 128, H_, B_);
  mha_attn_kernel<<<gattn, blk, 0, stream>>>(qws, kws, vws, xws);

  dim3 gout(N_ / 128, D_ / 64, B_);
  mha_out_kernel<<<gout, blk, 0, stream>>>(xws, Wm, bm, (float*)d_out);
}
